// MoE_regression_33148557590596
// MI455X (gfx1250) — matrix operand order repaired and checked by reading, not compiled, not run on hardware
//
#include <hip/hip_runtime.h>
#include <hip/hip_bf16.h>

// MoE top-2 regression, MI455X (gfx1250, wave32).
// Sparse routing + bf16 WMMA grouped GEMM (64-token M-blocks, 4x B reuse per wave,
// double-buffered B across k-steps + pipelined A fragments) + fused
// LN/ReLU/W2/sigmoid epilogue. ~220 live VGPRs < 256 budget: no scratch spills.

typedef __attribute__((ext_vector_type(16))) __bf16 v16bf;
typedef __attribute__((ext_vector_type(8)))  float  v8f;
typedef __attribute__((ext_vector_type(4)))  float  vf4;

#define B_TOK   16384
#define D_IN    1200
#define N_EXP   8
#define H_DIM   1024
#define KSTEPS  38      // ceil(1200/32), even -> clean unroll-by-2
#define KPAD    1216    // KSTEPS*32
#define TILE_M  64      // tokens per block (4 M-tiles of 16)
#define THREADS 512     // 16 waves; each wave: 4 M-tiles x 4 N-tiles (64 cols)

union Frag { v16bf v; vf4 f[2]; };

__device__ __forceinline__ unsigned short f2bf(float f) {
  unsigned u = __float_as_uint(f);
  u += 0x7FFFu + ((u >> 16) & 1u);   // round-to-nearest-even
  return (unsigned short)(u >> 16);
}

// ---------------------------------------------------------------- init
__global__ void init_kernel(float* __restrict__ y, int* __restrict__ cnt) {
  int i = blockIdx.x * blockDim.x + threadIdx.x;
  if (i < B_TOK) y[i] = 0.f;
  if (i < N_EXP) cnt[i] = 0;
}

// ------------------------------------------------- pack W1 -> bf16 WMMA B layout
// W1p[e][t][n][kk], kk in [0,32): K = 32t + 16*(kk/16) + (kk%16)
// => each lane's B fragment (16 bf16) is 32 contiguous bytes.
__global__ void pack_w1_kernel(const float* __restrict__ W1,
                               unsigned int* __restrict__ W1p) {
  int i = blockIdx.x * blockDim.x + threadIdx.x;  // one dword pair per thread
  const int total = N_EXP * KSTEPS * H_DIM * 16;
  if (i >= total) return;
  int kkp = i & 15;
  int n   = (i >> 4) & (H_DIM - 1);
  int r   = i >> 14;
  int t   = r % KSTEPS;
  int e   = r / KSTEPS;
  int kk0 = kkp * 2;                              // even -> pair stays in same half
  int K0  = 32 * t + (kk0 & 16) + (kk0 & 15);
  int K1  = K0 + 1;
  float f0 = (K0 < D_IN) ? W1[((long)e * D_IN + K0) * H_DIM + n] : 0.f;
  float f1 = (K1 < D_IN) ? W1[((long)e * D_IN + K1) * H_DIM + n] : 0.f;
  W1p[(((long)e * KSTEPS + t) * H_DIM + n) * 16 + kkp] =
      (unsigned int)f2bf(f0) | ((unsigned int)f2bf(f1) << 16);
}

// ---------------------------------------------------------------- gating
// One wave32 per token: logits = x @ w_gate, softmax, top-2, scatter to expert lists.
__global__ void gate_kernel(const float* __restrict__ x, const float* __restrict__ wg,
                            int* __restrict__ cnt, int* __restrict__ listTok,
                            float* __restrict__ listGate) {
  int tok  = blockIdx.x * (blockDim.x >> 5) + (threadIdx.x >> 5);
  int lane = threadIdx.x & 31;
  if (tok >= B_TOK) return;
  const float* xr = x + (long)tok * D_IN;
  float a[N_EXP];
  #pragma unroll
  for (int e = 0; e < N_EXP; ++e) a[e] = 0.f;
  for (int d = lane; d < D_IN; d += 32) {
    float xv = xr[d];
    const float* w = wg + d * N_EXP;
    #pragma unroll
    for (int e = 0; e < N_EXP; ++e) a[e] = fmaf(xv, w[e], a[e]);
  }
  #pragma unroll
  for (int m = 16; m >= 1; m >>= 1) {
    #pragma unroll
    for (int e = 0; e < N_EXP; ++e) a[e] += __shfl_xor(a[e], m, 32);
  }
  if (lane == 0) {
    float mx = a[0];
    #pragma unroll
    for (int e = 1; e < N_EXP; ++e) mx = fmaxf(mx, a[e]);
    float p[N_EXP]; float sum = 0.f;
    #pragma unroll
    for (int e = 0; e < N_EXP; ++e) { p[e] = __expf(a[e] - mx); sum += p[e]; }
    float inv = 1.f / sum;
    #pragma unroll
    for (int e = 0; e < N_EXP; ++e) p[e] *= inv;
    int i0 = 0; float v0 = p[0];
    #pragma unroll
    for (int e = 1; e < N_EXP; ++e) if (p[e] > v0) { v0 = p[e]; i0 = e; }
    int i1 = -1; float v1 = -1.f;
    #pragma unroll
    for (int e = 0; e < N_EXP; ++e) if (e != i0 && p[e] > v1) { v1 = p[e]; i1 = e; }
    float denom = v0 + v1 + 1e-6f;
    float g0 = v0 / denom, g1v = v1 / denom;
    int pos = atomicAdd(&cnt[i0], 1);
    listTok[i0 * B_TOK + pos] = tok; listGate[i0 * B_TOK + pos] = g0;
    pos = atomicAdd(&cnt[i1], 1);
    listTok[i1 * B_TOK + pos] = tok; listGate[i1 * B_TOK + pos] = g1v;
  }
}

// ------------------------------------------- grouped expert GEMM + fused epilogue
// Block: 512 threads = 16 waves; 64 tokens x 1024 H columns per block.
// Wave w owns columns [w*64, w*64+64) as 4 WMMA n-tiles and ALL 4 M-tiles:
// each B fragment feeds 4 WMMAs (4x L2 weight-traffic reuse).
// B double-buffered across k-steps (loads for t+1 in flight during t's WMMAs);
// A fragments pipelined one M-group ahead.
__global__ __launch_bounds__(THREADS) void moe_expert_kernel(
    const float* __restrict__ x, const unsigned short* __restrict__ W1p,
    const float* __restrict__ b1, const float* __restrict__ g1,
    const float* __restrict__ be1, const float* __restrict__ W2,
    const float* __restrict__ b2, const int* __restrict__ cnt,
    const int* __restrict__ listTok, const float* __restrict__ listGate,
    float* __restrict__ y) {
  __shared__ unsigned short As[TILE_M * KPAD];  // 155.6 KB bf16 A tile (gfx1250: 320KB/WG)
  __shared__ int   shTok[TILE_M];
  __shared__ float shGate[TILE_M];
  __shared__ float shSum[TILE_M], shSS[TILE_M], shMu[TILE_M], shRs[TILE_M], shDot[TILE_M];

  const int e     = blockIdx.y;
  const int tile  = blockIdx.x;
  const int count = cnt[e];
  if (tile * TILE_M >= count) return;           // uniform exit: EXEC stays full

  const int tid  = threadIdx.x;
  const int wid  = tid >> 5;                    // 0..15
  const int lane = tid & 31;
  const int half = lane >> 4;
  const int ln   = lane & 15;

  if (tid < TILE_M) {
    int idx = tile * TILE_M + tid;
    int src = (idx < count) ? idx : (count - 1);
    shTok[tid]  = listTok[e * B_TOK + src];
    shGate[tid] = (idx < count) ? listGate[e * B_TOK + src] : 0.f;
    shSum[tid] = 0.f; shSS[tid] = 0.f; shDot[tid] = 0.f;
  }
  __syncthreads();

  // Stage gathered token rows into LDS as bf16 (8 threads per row).
  {
    int r  = tid >> 3;       // 0..63
    int c0 = tid & 7;
    const float* xr = x + (long)shTok[r] * D_IN;
    for (int c = c0; c < KPAD; c += 8) {
      float v = (c < D_IN) ? xr[c] : 0.f;
      As[r * KPAD + c] = f2bf(v);
    }
  }
  __syncthreads();

  // Per-wave bases; per-(t,nt,mg) addresses are immediate offsets off these.
  const unsigned short* Wcol =
      W1p + (long)e * KSTEPS * (H_DIM * 32) + ((long)wid * 64 + ln) * 32 + 16 * half;
  const v8f vzero = {0.f, 0.f, 0.f, 0.f, 0.f, 0.f, 0.f, 0.f};
  v8f acc[4][4];                                 // [M-group][N-tile], 128 VGPRs
  #pragma unroll
  for (int mg = 0; mg < 4; ++mg)
    #pragma unroll
    for (int nt = 0; nt < 4; ++nt) acc[mg][nt] = vzero;

  // Load all 4 B fragments of k-step t (8 b128 loads, immediate nt offsets).
  auto loadB = [&](Frag (&bb)[4], int t) {
    const unsigned short* bbase = Wcol + (long)t * (H_DIM * 32);
    #pragma unroll
    for (int nt = 0; nt < 4; ++nt) {             // nt stride = 16 cols * 32 = 512 elems
      bb[nt].f[0] = *reinterpret_cast<const vf4*>(bbase + nt * 512);
      bb[nt].f[1] = *reinterpret_cast<const vf4*>(bbase + nt * 512 + 8);
    }
    __builtin_prefetch(bbase + H_DIM * 32, 0, 1);  // speculative: next k-step
  };

  // 16 WMMAs of one k-step; A fragments pipelined one M-group ahead.
  auto kcompute = [&](const Frag (&bb)[4], int t) {
    Frag a0, a1;
    {
      const unsigned short* ap = &As[ln * KPAD + 32 * t + 16 * half];
      a0.f[0] = *reinterpret_cast<const vf4*>(ap);
      a0.f[1] = *reinterpret_cast<const vf4*>(ap + 8);
    }
    #pragma unroll
    for (int mg = 0; mg < 4; ++mg) {
      Frag& acur = (mg & 1) ? a1 : a0;
      Frag& anxt = (mg & 1) ? a0 : a1;
      if (mg < 3) {
        const unsigned short* ap = &As[((mg + 1) * 16 + ln) * KPAD + 32 * t + 16 * half];
        anxt.f[0] = *reinterpret_cast<const vf4*>(ap);
        anxt.f[1] = *reinterpret_cast<const vf4*>(ap + 8);
      }
      #pragma unroll
      for (int nt = 0; nt < 4; ++nt) {
        acc[mg][nt] = __builtin_amdgcn_wmma_f32_16x16x32_bf16(
            false, acur.v, false, bb[nt].v, (short)0, acc[mg][nt], false, false);
      }
    }
  };

  Frag bA[4], bB[4];
  loadB(bA, 0);
  #pragma clang loop unroll(disable)
  for (int t = 0; t < KSTEPS; t += 2) {
    loadB(bB, t + 1);                            // in flight during t's WMMAs
    kcompute(bA, t);
    loadB(bA, (t + 2 < KSTEPS) ? (t + 2) : 0);   // clamped dummy on last iter
    kcompute(bB, t + 1);
  }

  // ---- epilogue: +b1, LayerNorm over H, ReLU, dot W2[:,0], sigmoid, gated combine
  // C/D layout: acc[mg][nt][j] -> row = 16*mg + j + 8*half, col = wid*64 + nt*16 + ln
  #pragma unroll
  for (int mg = 0; mg < 4; ++mg) {
    float s[8], ss[8];
    #pragma unroll
    for (int j = 0; j < 8; ++j) { s[j] = 0.f; ss[j] = 0.f; }
    #pragma unroll
    for (int nt = 0; nt < 4; ++nt) {
      int col = wid * 64 + nt * 16 + ln;
      float b1v = b1[e * H_DIM + col];
      #pragma unroll
      for (int j = 0; j < 8; ++j) {
        float c = acc[mg][nt][j] + b1v;
        acc[mg][nt][j] = c;
        s[j] += c;
        ss[j] = fmaf(c, c, ss[j]);
      }
    }
    #pragma unroll
    for (int m = 8; m >= 1; m >>= 1) {
      #pragma unroll
      for (int j = 0; j < 8; ++j) {
        s[j]  += __shfl_xor(s[j],  m, 16);
        ss[j] += __shfl_xor(ss[j], m, 16);
      }
    }
    if (ln == 0) {
      #pragma unroll
      for (int j = 0; j < 8; ++j) {
        atomicAdd(&shSum[16 * mg + j + 8 * half], s[j]);
        atomicAdd(&shSS [16 * mg + j + 8 * half], ss[j]);
      }
    }
  }
  __syncthreads();
  if (tid < TILE_M) {
    float mu  = shSum[tid] * (1.f / H_DIM);
    float var = shSS[tid] * (1.f / H_DIM) - mu * mu;
    shMu[tid] = mu;
    shRs[tid] = rsqrtf(var + 1e-5f);
  }
  __syncthreads();

  #pragma unroll
  for (int mg = 0; mg < 4; ++mg) {
    float p[8];
    #pragma unroll
    for (int j = 0; j < 8; ++j) p[j] = 0.f;
    #pragma unroll
    for (int nt = 0; nt < 4; ++nt) {
      int col = wid * 64 + nt * 16 + ln;
      float g1v = g1 [e * H_DIM + col];
      float bev = be1[e * H_DIM + col];
      float w2v = W2 [e * H_DIM + col];
      #pragma unroll
      for (int j = 0; j < 8; ++j) {
        int row = 16 * mg + j + 8 * half;
        float hn = (acc[mg][nt][j] - shMu[row]) * shRs[row] * g1v + bev;
        hn = fmaxf(hn, 0.f);
        p[j] = fmaf(hn, w2v, p[j]);
      }
    }
    #pragma unroll
    for (int m = 8; m >= 1; m >>= 1) {
      #pragma unroll
      for (int j = 0; j < 8; ++j) p[j] += __shfl_xor(p[j], m, 16);
    }
    if (ln == 0) {
      #pragma unroll
      for (int j = 0; j < 8; ++j) atomicAdd(&shDot[16 * mg + j + 8 * half], p[j]);
    }
  }
  __syncthreads();
  if (tid < TILE_M && tile * TILE_M + tid < count) {
    float o = shDot[tid] + b2[e];
    o = 1.f / (1.f + __expf(-o));
    atomicAdd(&y[shTok[tid]], shGate[tid] * o);   // exactly 2 adds/token: deterministic
  }
}

// ---------------------------------------------------------------- launch
extern "C" void kernel_launch(void* const* d_in, const int* in_sizes, int n_in,
                              void* d_out, int out_size, void* d_ws, size_t ws_size,
                              hipStream_t stream) {
  const float* x   = (const float*)d_in[0];
  const float* wg  = (const float*)d_in[1];
  const float* W1  = (const float*)d_in[2];
  const float* b1  = (const float*)d_in[3];
  const float* g1  = (const float*)d_in[4];
  const float* be1 = (const float*)d_in[5];
  const float* W2  = (const float*)d_in[6];
  const float* b2  = (const float*)d_in[7];
  float* y = (float*)d_out;

  char* ws = (char*)d_ws;
  int*   cnt      = (int*)ws;   ws += 256;
  int*   listTok  = (int*)ws;   ws += (size_t)N_EXP * B_TOK * sizeof(int);
  float* listGate = (float*)ws; ws += (size_t)N_EXP * B_TOK * sizeof(float);
  unsigned short* W1p = (unsigned short*)ws;  // ~20 MB packed bf16 weights

  init_kernel<<<(B_TOK + 255) / 256, 256, 0, stream>>>(y, cnt);

  const int packTotal = N_EXP * KSTEPS * H_DIM * 16;
  pack_w1_kernel<<<(packTotal + 255) / 256, 256, 0, stream>>>(W1, (unsigned int*)W1p);

  gate_kernel<<<B_TOK / 8, 256, 0, stream>>>(x, wg, cnt, listTok, listGate);

  dim3 grid(B_TOK / TILE_M, N_EXP);   // blocks past each expert's count exit early
  moe_expert_kernel<<<grid, THREADS, 0, stream>>>(x, W1p, b1, g1, be1, W2, b2,
                                                  cnt, listTok, listGate, y);
}
